// Attention_1116691497608
// MI455X (gfx1250) — matrix operand order repaired and checked by reading, not compiled, not run on hardware
//
#include <hip/hip_runtime.h>

// ---------------------------------------------------------------------------
// CDNA5 (gfx1250) attention layer: QKV proj (bf16 WMMA) -> RMSNorm+RoPE3D ->
// flash attention (bf16 WMMA, online softmax, async global->LDS V staging) ->
// output proj (bf16 WMMA). All matmuls: v_wmma_f32_16x16x32_bf16 (f32 accum).
// ---------------------------------------------------------------------------

typedef __attribute__((ext_vector_type(16))) __bf16 v16bf;
typedef __attribute__((ext_vector_type(8)))  __bf16 bf8;
typedef __attribute__((ext_vector_type(8)))  float  v8f;

union Frag16 { v16bf v; bf8 h2[2]; __bf16 e[16]; };

__device__ __forceinline__ v8f zero8() {
  v8f z;
#pragma unroll
  for (int i = 0; i < 8; ++i) z[i] = 0.0f;
  return z;
}

__device__ __forceinline__ v8f wmma_bf16(const Frag16& a, const Frag16& b, v8f c) {
  // 8 args: (neg_a, A, neg_b, B, c_mod, C, reuse_a, reuse_b)
  return __builtin_amdgcn_wmma_f32_16x16x32_bf16(false, a.v, false, b.v,
                                                 (short)0, c, false, false);
}

// ---------------------------------------------------------------------------
// f32 -> bf16 convert with zero row-padding (rows..rowsPad-1 set to 0).
// ---------------------------------------------------------------------------
__global__ void k_cvt_pad(const float* __restrict__ src, __bf16* __restrict__ dst,
                          int rows, int cols, int rowsPad) {
  long i = (long)blockIdx.x * blockDim.x + threadIdx.x;
  long tot = (long)rowsPad * cols;
  if (i >= tot) return;
  int r = (int)(i / cols);
  float v = (r < rows) ? src[i] : 0.0f;
  dst[i] = (__bf16)v;
}

// ---------------------------------------------------------------------------
// out[M,Cout] = A[M,K](bf16) @ W[Cout,K]^T (bf16) + bias, f32 out.
// Block = 128 threads = 4 waves; wave computes 16(M) x 64(N), reusing one
// A-fragment across 4 B-fragments per 32-wide K step.
// Fragment layouts used by this kernel (contiguous K, same for A and B):
//   A 16x32 bf16: lane row = lane&15; elems e -> K = e + 16*(lane>=16)
//   B 32x16 bf16: lane col = lane&15; elems e -> K = e + 16*(lane>=16)
//   C 16x16 f32 : reg r, lane -> row = r + 8*(lane>=16), col = lane&15
// ---------------------------------------------------------------------------
__global__ void k_gemm_bias(const __bf16* __restrict__ A, const __bf16* __restrict__ W,
                            const float* __restrict__ bias, float* __restrict__ out,
                            int K, int rowsStore, int ldOut) {
  const int lane = threadIdx.x & 31;
  const int wv   = threadIdx.x >> 5;
  const int li   = lane & 15;
  const int hf   = lane >> 4;
  const int Mb   = blockIdx.y * 64 + wv * 16;
  const int Nb   = blockIdx.x * 64;

  v8f acc[4];
#pragma unroll
  for (int j = 0; j < 4; ++j) acc[j] = zero8();

  const __bf16* arow = A + (size_t)(Mb + li) * K;
  const int nsteps = K >> 5;
  for (int kk = 0; kk < nsteps; ++kk) {
    Frag16 af;
    const __bf16* ap = arow + kk * 32 + 16 * hf;
    // prefetch the activation stream ~1KB ahead (-> global_prefetch_b8)
    __builtin_prefetch((const void*)(ap + 512), 0, 1);
    af.h2[0] = *(const bf8*)(ap);        // K = kk*32 + 16*hf + [0..7]
    af.h2[1] = *(const bf8*)(ap + 8);    // K = kk*32 + 16*hf + [8..15]
#pragma unroll
    for (int j = 0; j < 4; ++j) {
      Frag16 bfr;
      const __bf16* bp = W + (size_t)(Nb + j * 16 + li) * K + kk * 32 + 16 * hf;
      bfr.h2[0] = *(const bf8*)(bp);     // K = kk*32 + 16*hf + [0..7]
      bfr.h2[1] = *(const bf8*)(bp + 8); // K = kk*32 + 16*hf + [8..15]
      acc[j] = wmma_bf16(af, bfr, acc[j]);
    }
  }

#pragma unroll
  for (int j = 0; j < 4; ++j) {
    const int col = Nb + j * 16 + li;
    const float bv = bias[col];
#pragma unroll
    for (int r = 0; r < 8; ++r) {
      const int row = Mb + r + 8 * hf;
      if (row < rowsStore) out[(size_t)row * ldOut + col] = acc[j][r] + bv;
    }
  }
}

// ---------------------------------------------------------------------------
// Per (token n, head h): optional RMS-norm (Dh=96) + 3D RoPE, relayout from
// (Np, C) f32 to head-major (H, Np, 96) bf16. One wave per (n, h); each lane
// handles d = {lane, lane+32, lane+64} (one element per 32-wide RoPE chunk).
// Rotate-half within a chunk is exactly lane ^ 16.
// ---------------------------------------------------------------------------
__global__ void k_rms_rope(const float* __restrict__ src, const float* __restrict__ wrms,
                           __bf16* __restrict__ dst, int N, int Np, int C,
                           const int* __restrict__ TTp, int doRope) {
  constexpr int DH = 96;
  const int n    = blockIdx.x;
  const int h    = blockIdx.y;
  const int lane = threadIdx.x;

  const float* row = src + (size_t)n * C + h * DH;
  float x0 = row[lane], x1 = row[lane + 32], x2 = row[lane + 64];

  if (doRope) {
    // RMS over the 96-wide head dim
    float ss = x0 * x0 + x1 * x1 + x2 * x2;
#pragma unroll
    for (int off = 1; off < 32; off <<= 1) ss += __shfl_xor(ss, off, 32);
    const float rinv = rsqrtf(ss * (1.0f / DH) + 1e-6f);
    x0 = x0 * rinv * wrms[lane];
    x1 = x1 * rinv * wrms[lane + 32];
    x2 = x2 * rinv * wrms[lane + 64];

    const int TT = *TTp;
    if (n >= TT && n < N) {
      // (t,h,w) grid per the reference's _thw table
      const int vid = N - TT;
      int hd = 23, wd = 40;
      if      (vid == 2640) { hd = 22; wd = 40; }
      else if (vid == 1530) { hd = 17; wd = 30; }
      else if (vid == 6120) { hd = 34; wd = 60; }
      else if (vid == 660)  { hd = 11; wd = 20; }
      const int m = n - TT;
      float pos[3];
      pos[0] = (float)(m / (hd * wd));
      pos[1] = (float)((m / wd) % hd);
      pos[2] = (float)(m % wd);

      const int fi = lane & 15;                       // frequency index in chunk
      const float inv = __powf(10000.0f, -(float)fi / 16.0f);
      float xs[3] = {x0, x1, x2};
#pragma unroll
      for (int j = 0; j < 3; ++j) {
        const float ang = pos[j] * inv;
        const float c = __cosf(ang), s = __sinf(ang);
        const float partner = __shfl_xor(xs[j], 16, 32);
        xs[j] = (lane < 16) ? (xs[j] * c - partner * s)   // out[j]    = x1*cos - x2*sin
                            : (xs[j] * c + partner * s);  // out[j+16] = x2*cos + x1*sin
      }
      x0 = xs[0]; x1 = xs[1]; x2 = xs[2];
    }
  }
  if (n >= N) { x0 = 0.0f; x1 = 0.0f; x2 = 0.0f; }     // zero the pad rows

  __bf16* drow = dst + ((size_t)h * Np + n) * DH;
  drow[lane]      = (__bf16)x0;
  drow[lane + 32] = (__bf16)x1;
  drow[lane + 64] = (__bf16)x2;
}

// ---------------------------------------------------------------------------
// Flash attention, one wave per (32-query tile, head). q/k/v head-major
// (H, Np, 96) bf16. Two 16-row q tiles share all K/V fragment loads.
// Per 32-key block:
//   1) async-stage the 32x96 bf16 V tile into LDS (global_load_async_to_lds,
//      ASYNCcnt) -- overlapped with score WMMAs + softmax,
//   2) 12 score WMMAs (2 q tiles x 2 key cols x 3 K-steps), pad-mask,
//      online softmax with row stats per (reg, half),
//   3) P staged through LDS (C-layout -> A-layout transpose),
//   4) s_wait_asynccnt 0, build V B-fragments from LDS, 12 P@V WMMAs.
// ---------------------------------------------------------------------------
__global__ void k_attn(const __bf16* __restrict__ qb, const __bf16* __restrict__ kb,
                       const __bf16* __restrict__ vb, __bf16* __restrict__ ob,
                       int N, int Np, int C) {
  constexpr int DH = 96;
  __shared__ __attribute__((aligned(16))) __bf16 Vld[32 * DH];     // 6 KB
  __shared__ __attribute__((aligned(16))) __bf16 Pld[2][16 * 32];  // 2 KB

  const int h    = blockIdx.y;
  const int qt   = blockIdx.x;                 // 32-query tile index
  const int lane = threadIdx.x;
  const int li   = lane & 15;
  const int hf   = lane >> 4;

  const __bf16* qh = qb + (size_t)h * Np * DH;
  const __bf16* kh = kb + (size_t)h * Np * DH;
  const __bf16* vh = vb + (size_t)h * Np * DH;

  // Q fragments: 2 row tiles x 3 K-steps (d = 0..31, 32..63, 64..95)
  Frag16 qA[2][3];
#pragma unroll
  for (int u = 0; u < 2; ++u) {
    const __bf16* qp = qh + (size_t)(qt * 32 + u * 16 + li) * DH;
#pragma unroll
    for (int t = 0; t < 3; ++t) {
      qA[u][t].h2[0] = *(const bf8*)(qp + 32 * t + 16 * hf);
      qA[u][t].h2[1] = *(const bf8*)(qp + 32 * t + 16 * hf + 8);
    }
  }

  v8f acc[2][6];
  float m[2][8], l[2][8];
#pragma unroll
  for (int u = 0; u < 2; ++u) {
#pragma unroll
    for (int dt = 0; dt < 6; ++dt) acc[u][dt] = zero8();
#pragma unroll
    for (int r = 0; r < 8; ++r) { m[u][r] = -3.0e38f; l[u][r] = 0.0f; }
  }

  const float scale = rsqrtf((float)DH);
  const unsigned ldsV = (unsigned)(size_t)&Vld[0];   // LDS byte offset (addr[31:0])
  const int nkb = Np >> 5;

  for (int kb2 = 0; kb2 < nkb; ++kb2) {
    // ---- 1) async stage of the V tile (32 keys x 96 d, row-major, 6144 B) ----
    {
      const unsigned long long gV =
          (unsigned long long)(size_t)(vh + (size_t)(kb2 * 32) * DH);
#pragma unroll
      for (int i = 0; i < 12; ++i) {
        const unsigned off = (unsigned)(i * 512 + lane * 16);
        unsigned       lo  = ldsV + off;
        unsigned long long ga = gV + off;
        asm volatile("global_load_async_to_lds_b128 %0, %1, off"
                     :: "v"(lo), "v"(ga) : "memory");
      }
    }

    // ---- 2) scores S = q @ k^T for 32 keys, both q tiles ----
    v8f s[2][2];
#pragma unroll
    for (int u = 0; u < 2; ++u) { s[u][0] = zero8(); s[u][1] = zero8(); }
    const __bf16* kp0 = kh + (size_t)(kb2 * 32 + li) * DH;
    const __bf16* kp1 = kp0 + (size_t)16 * DH;
#pragma unroll
    for (int t = 0; t < 3; ++t) {
      Frag16 b0, b1;
      b0.h2[0] = *(const bf8*)(kp0 + 32 * t + 16 * hf);
      b0.h2[1] = *(const bf8*)(kp0 + 32 * t + 16 * hf + 8);
      b1.h2[0] = *(const bf8*)(kp1 + 32 * t + 16 * hf);
      b1.h2[1] = *(const bf8*)(kp1 + 32 * t + 16 * hf + 8);
#pragma unroll
      for (int u = 0; u < 2; ++u) {
        s[u][0] = wmma_bf16(qA[u][t], b0, s[u][0]);
        s[u][1] = wmma_bf16(qA[u][t], b1, s[u][1]);
      }
    }

    // ---- online softmax over this 32-key block, per q tile ----
    const bool ok0 = (kb2 * 32 + li) < N;
    const bool ok1 = (kb2 * 32 + 16 + li) < N;
#pragma unroll
    for (int u = 0; u < 2; ++u) {
      float p0[8], p1[8];
#pragma unroll
      for (int r = 0; r < 8; ++r) {
        float a0 = ok0 ? s[u][0][r] * scale : -3.0e38f;
        float a1 = ok1 ? s[u][1][r] * scale : -3.0e38f;
        float bm = fmaxf(a0, a1);
#pragma unroll
        for (int off = 1; off < 16; off <<= 1) bm = fmaxf(bm, __shfl_xor(bm, off, 16));
        const float mn = fmaxf(m[u][r], bm);
        const float al = __expf(m[u][r] - mn);
        m[u][r] = mn;
        l[u][r] *= al;
#pragma unroll
        for (int dt = 0; dt < 6; ++dt) acc[u][dt][r] *= al;
        const float e0 = __expf(a0 - mn);
        const float e1 = __expf(a1 - mn);
        p0[r] = e0; p1[r] = e1;
        float rs = e0 + e1;
#pragma unroll
        for (int off = 1; off < 16; off <<= 1) rs += __shfl_xor(rs, off, 16);
        l[u][r] += rs;
      }
      // ---- 3) P (C layout) -> LDS ----
#pragma unroll
      for (int r = 0; r < 8; ++r) {
        const int row = r + 8 * hf;
        Pld[u][row * 32 + li]      = (__bf16)p0[r];
        Pld[u][row * 32 + 16 + li] = (__bf16)p1[r];
      }
    }
    __syncthreads();
    Frag16 pa[2];
#pragma unroll
    for (int u = 0; u < 2; ++u) {
      pa[u].h2[0] = *(const bf8*)&Pld[u][li * 32 + 16 * hf];
      pa[u].h2[1] = *(const bf8*)&Pld[u][li * 32 + 16 * hf + 8];
    }

    // ---- 4) wait async V, then O += P @ V (K = 32 keys) ----
    asm volatile("s_wait_asynccnt 0x0" ::: "memory");
#pragma unroll
    for (int dt = 0; dt < 6; ++dt) {
      Frag16 vB;
#pragma unroll
      for (int e = 0; e < 16; ++e)
        vB.e[e] = Vld[(e + 16 * hf) * DH + dt * 16 + li];
#pragma unroll
      for (int u = 0; u < 2; ++u)
        acc[u][dt] = wmma_bf16(pa[u], vB, acc[u][dt]);
    }
    __syncthreads();
  }

  // ---- normalize and store O as (Np, C) bf16, row-major ----
#pragma unroll
  for (int u = 0; u < 2; ++u) {
#pragma unroll
    for (int dt = 0; dt < 6; ++dt) {
#pragma unroll
      for (int r = 0; r < 8; ++r) {
        const int row = r + 8 * hf;
        const int n = qt * 32 + u * 16 + row;
        const float val = acc[u][dt][r] / l[u][r];
        ob[(size_t)n * C + h * DH + dt * 16 + li] = (__bf16)val;
      }
    }
  }
}

// ---------------------------------------------------------------------------
// Host side
// ---------------------------------------------------------------------------
extern "C" void kernel_launch(void* const* d_in, const int* in_sizes, int n_in,
                              void* d_out, int out_size, void* d_ws, size_t ws_size,
                              hipStream_t stream) {
  const float* x  = (const float*)d_in[0];
  const float* Wq = (const float*)d_in[1];
  const float* bq = (const float*)d_in[2];
  const float* Wk = (const float*)d_in[3];
  const float* bk = (const float*)d_in[4];
  const float* Wv = (const float*)d_in[5];
  const float* bv = (const float*)d_in[6];
  const float* qn = (const float*)d_in[7];
  const float* kn = (const float*)d_in[8];
  const float* Wp = (const float*)d_in[9];
  const float* bp = (const float*)d_in[10];
  const int*   TT = (const int*)d_in[11];

  const int C  = in_sizes[2];               // bq length = 1536
  const int N  = in_sizes[0] / C;           // 2866
  const int H  = 16;
  const int Np = ((N + 63) / 64) * 64;      // 2880 (tile-aligned, zero-padded)

  // Workspace carve-up (~116 MB total for the reference sizes)
  char* w = (char*)d_ws;
  auto alloc = [&](size_t bytes) -> char* {
    char* p = w;
    w += (bytes + 255) & ~(size_t)255;
    return p;
  };
  const size_t NpC = (size_t)Np * C;
  const size_t CC  = (size_t)C * C;
  __bf16* xb  = (__bf16*)alloc(NpC * 2);
  __bf16* wqb = (__bf16*)alloc(CC * 2);
  __bf16* wkb = (__bf16*)alloc(CC * 2);
  __bf16* wvb = (__bf16*)alloc(CC * 2);
  __bf16* wpb = (__bf16*)alloc(CC * 2);
  float*  qf  = (float*)alloc(NpC * 4);
  float*  kf  = (float*)alloc(NpC * 4);
  float*  vf  = (float*)alloc(NpC * 4);
  __bf16* qbb = (__bf16*)alloc(NpC * 2);
  __bf16* kbb = (__bf16*)alloc(NpC * 2);
  __bf16* vbb = (__bf16*)alloc(NpC * 2);
  __bf16* obb = (__bf16*)alloc(NpC * 2);
  (void)n_in; (void)out_size; (void)ws_size;

  const int th = 256;
  k_cvt_pad<<<(unsigned)((NpC + th - 1) / th), th, 0, stream>>>(x,  xb,  N, C, Np);
  k_cvt_pad<<<(unsigned)((CC  + th - 1) / th), th, 0, stream>>>(Wq, wqb, C, C, C);
  k_cvt_pad<<<(unsigned)((CC  + th - 1) / th), th, 0, stream>>>(Wk, wkb, C, C, C);
  k_cvt_pad<<<(unsigned)((CC  + th - 1) / th), th, 0, stream>>>(Wv, wvb, C, C, C);
  k_cvt_pad<<<(unsigned)((CC  + th - 1) / th), th, 0, stream>>>(Wp, wpb, C, C, C);

  dim3 gg(C / 64, Np / 64);
  k_gemm_bias<<<gg, 128, 0, stream>>>(xb, wqb, bq, qf, C, Np, C);
  k_gemm_bias<<<gg, 128, 0, stream>>>(xb, wkb, bk, kf, C, Np, C);
  k_gemm_bias<<<gg, 128, 0, stream>>>(xb, wvb, bv, vf, C, Np, C);

  dim3 gr(Np, H);
  k_rms_rope<<<gr, 32, 0, stream>>>(qf, qn, qbb, N, Np, C, TT, 1);
  k_rms_rope<<<gr, 32, 0, stream>>>(kf, kn, kbb, N, Np, C, TT, 1);
  k_rms_rope<<<gr, 32, 0, stream>>>(vf, qn, vbb, N, Np, C, TT, 0);  // wrms unused

  dim3 ga(Np / 32, H);
  k_attn<<<ga, 32, 0, stream>>>(qbb, kbb, vbb, obb, N, Np, C);

  k_gemm_bias<<<gg, 128, 0, stream>>>(obb, wpb, bp, (float*)d_out, C, N, C);
}
